// OpInfs_torch_50062138802770
// MI455X (gfx1250) — matrix operand order repaired and checked by reading, not compiled, not run on hardware
//
#include <hip/hip_runtime.h>

// ---------------------------------------------------------------------------
// out[128,16384] = A@x + B@x2,  x2 = triu pairwise products of x rows.
// Fused as out = W @ F,  W = [A|B] (128 x 8384),  F = [x ; x2] built on the fly.
// bf16x2-split GEMM on v_wmma_f32_16x16x32_bf16 (near-fp32 accuracy, ~8x f32
// matrix rate, 3 WMMAs per 32-K tile).  Software-pipelined: F tile is
// double-buffered so feature building (VALU+LDS) co-executes with WMMA.
// ---------------------------------------------------------------------------

typedef __attribute__((ext_vector_type(16))) __bf16 v16bf;
typedef __attribute__((ext_vector_type(8)))  float  v8f;

#define NROWS   128
#define NBATCH  16384
#define KTOT    8384          // 128 (A) + 8256 (B)
#define KSTEPS  262           // KTOT / 32
#define KC      64            // K-chunk staged in LDS per iteration
#define NCHUNK  131           // KTOT / KC
#define FSTR    80            // F tile row stride in bf16 elems (160B: 16B-aligned, bank-spread)
#define WPK_ELEMS (NROWS * KTOT)          // 1,073,152 bf16 per (hi|lo) plane
#define WS_RC_OFF 0
#define WS_WH_OFF 65536
#define WS_WL_OFF (65536 + 2146304)       // WPK_ELEMS * 2 bytes

// ---- bf16 helpers ----------------------------------------------------------
__device__ __forceinline__ unsigned short bf16_rne(float f) {
    unsigned u = __builtin_bit_cast(unsigned, f);
    unsigned r = u + 0x7FFFu + ((u >> 16) & 1u);
    return (unsigned short)(r >> 16);
}

__device__ __forceinline__ void split_bf16(float f, unsigned short& hi, unsigned short& lo) {
    hi = bf16_rne(f);
    float fh = __builtin_bit_cast(float, (unsigned)hi << 16);
    lo = bf16_rne(f - fh);
}

struct bytes32 { uint4 a, b; };
__device__ __forceinline__ v16bf as_v16bf(uint4 a, uint4 b) {
    bytes32 t{a, b};
    return __builtin_bit_cast(v16bf, t);
}

// ---- prep 1: triu index table. rc[k] = (row<<16)|col; col==0xFFFF => F=x[row]
__global__ void build_rc_kernel(unsigned int* __restrict__ rc) {
    const int i = blockIdx.x;                       // 0..127
    if (threadIdx.x == 0) rc[i] = ((unsigned)i << 16) | 0xFFFFu;
    const int off = NROWS + i * NROWS - (i * (i - 1)) / 2;
    for (int j = i + threadIdx.x; j < NROWS; j += blockDim.x)
        rc[off + (j - i)] = ((unsigned)i << 16) | (unsigned)j;
}

// ---- prep 2: split W=[A|B] into hi/lo bf16, pre-swizzled into the exact
// per-lane WMMA A-operand order so the main kernel's loads are coalesced b128s.
// flat layout: (((mt*KSTEPS + t)*2 + v)*32 + lane)*8 + e
__global__ __launch_bounds__(256)
void pack_w_kernel(const float* __restrict__ A, const float* __restrict__ B,
                   unsigned short* __restrict__ wh, unsigned short* __restrict__ wl) {
    const int fid = blockIdx.x * 256 + threadIdx.x;     // exact: 4192*256 == WPK_ELEMS
    const int e    = fid & 7;
    const int lane = (fid >> 3) & 31;
    const int v    = (fid >> 8) & 1;
    const int rest = fid >> 9;                          // mt*KSTEPS + t
    const int t    = rest % KSTEPS;
    const int mt   = rest / KSTEPS;
    const int h    = lane >> 4;
    const int E    = v * 8 + e;
    const int K    = t * 32 + (E < 8 ? E + 8 * h : E + 8 + 8 * h);   // ISA A layout
    const int m    = mt * 16 + (lane & 15);
    const float val = (K < NROWS) ? A[m * NROWS + K]
                                  : B[(long)m * 8256 + (K - NROWS)];
    unsigned short hi, lo;
    split_bf16(val, hi, lo);
    wh[fid] = hi;
    wl[fid] = lo;
}

// ---- main fused GEMM, software-pipelined ----------------------------------
__global__ __launch_bounds__(256)
void infs_main_kernel(const float* __restrict__ x,
                      const unsigned int* __restrict__ rc,
                      const uint4* __restrict__ wh,
                      const uint4* __restrict__ wl,
                      float* __restrict__ out) {
    __shared__ __align__(16) float          Xs[NROWS * 64];       // 32 KB f32 x tile
    __shared__ __align__(16) unsigned short FhS[2][64 * FSTR];    // 2 x 10 KB hi features
    __shared__ __align__(16) unsigned short FlS[2][64 * FSTR];    // 2 x 10 KB lo features

    const int tid  = threadIdx.x;
    const int c0   = blockIdx.x * 64;          // batch-column base of this block
    const int mt   = tid >> 5;                 // wave id == M tile (0..7)
    const int lane = tid & 31;
    const int h    = lane >> 4;
    const int cl   = lane & 15;

    // stage x tile: 128 rows x 64 cols, float4 coalesced
    for (int i = tid; i < NROWS * 16; i += 256) {
        const int row = i >> 4, q = i & 15;
        const float4 v = *(const float4*)&x[(long)row * NBATCH + c0 + q * 4];
        *(float4*)&Xs[row * 64 + q * 4] = v;
    }
    __syncthreads();

    const int fcol = tid & 63;                 // feature-build column
    const int kb   = tid >> 6;                 // feature-build k phase (0..3)

    // prologue: build chunk 0 into buffer 0
    {
        unsigned short* __restrict__ fh = &FhS[0][0];
        unsigned short* __restrict__ fl = &FlS[0][0];
#pragma unroll
        for (int ii = 0; ii < KC / 4; ++ii) {
            const int kk = kb + ii * 4;
            const unsigned p = rc[kk];
            const int r = p >> 16, c = p & 0xFFFF;
            float f = Xs[r * 64 + fcol];
            if (c != 0xFFFF) f *= Xs[c * 64 + fcol];
            unsigned short hi, lo;
            split_bf16(f, hi, lo);
            fh[fcol * FSTR + kk] = hi;
            fl[fcol * FSTR + kk] = lo;
        }
    }
    __syncthreads();

    v8f acc[4];
#pragma unroll
    for (int tl = 0; tl < 4; ++tl)
        acc[tl] = v8f{0.f, 0.f, 0.f, 0.f, 0.f, 0.f, 0.f, 0.f};

    for (int ch = 0; ch < NCHUNK; ++ch) {
        const int pb = ch & 1;
        const unsigned short* __restrict__ fhc = &FhS[pb][0];
        const unsigned short* __restrict__ flc = &FlS[pb][0];

        // -- consume chunk ch: 2 K-steps of 32, 4 column tiles, 3 WMMAs each
#pragma unroll
        for (int ks = 0; ks < KC / 32; ++ks) {
            const int t = ch * (KC / 32) + ks;
            const long wbase = ((long)(mt * KSTEPS + t) * 2) * 32 + lane;
            const uint4 a0h = wh[wbase], a1h = wh[wbase + 32];
            const uint4 a0l = wl[wbase], a1l = wl[wbase + 32];
            const v16bf ah = as_v16bf(a0h, a1h);
            const v16bf al = as_v16bf(a0l, a1l);
#pragma unroll
            for (int tl = 0; tl < 4; ++tl) {
                const int col = tl * 16 + cl;
                const int off = col * FSTR + ks * 32 + h * 8;    // K = E + 8h (E<8), E + 8 + 8h (E>=8)
                const uint4 b0h = *(const uint4*)&fhc[off];
                const uint4 b1h = *(const uint4*)&fhc[off + 16];
                const uint4 b0l = *(const uint4*)&flc[off];
                const uint4 b1l = *(const uint4*)&flc[off + 16];
                const v16bf bh = as_v16bf(b0h, b1h);
                const v16bf bl = as_v16bf(b0l, b1l);
                acc[tl] = __builtin_amdgcn_wmma_f32_16x16x32_bf16(
                    false, ah, false, bh, (short)0, acc[tl], false, false);
                acc[tl] = __builtin_amdgcn_wmma_f32_16x16x32_bf16(
                    false, ah, false, bl, (short)0, acc[tl], false, false);
                acc[tl] = __builtin_amdgcn_wmma_f32_16x16x32_bf16(
                    false, al, false, bh, (short)0, acc[tl], false, false);
            }
        }

        // -- build chunk ch+1 into the other buffer; co-executes with the
        //    WMMAs above (no barrier between them; buffer reuse is 2 barriers away)
        if (ch + 1 < NCHUNK) {
            unsigned short* __restrict__ fhn = &FhS[pb ^ 1][0];
            unsigned short* __restrict__ fln = &FlS[pb ^ 1][0];
            const int k0 = (ch + 1) * KC;
#pragma unroll
            for (int ii = 0; ii < KC / 4; ++ii) {
                const int kk = kb + ii * 4;
                const unsigned p = rc[k0 + kk];
                const int r = p >> 16, c = p & 0xFFFF;
                float f = Xs[r * 64 + fcol];
                if (c != 0xFFFF) f *= Xs[c * 64 + fcol];
                unsigned short hi, lo;
                split_bf16(f, hi, lo);
                fhn[fcol * FSTR + kk] = hi;
                fln[fcol * FSTR + kk] = lo;
            }
        }
        __syncthreads();
    }

    // -- write out: C/D layout: VGPR d -> row mt*16 + d + 8h, col = cl (ISA §7.12.2)
#pragma unroll
    for (int tl = 0; tl < 4; ++tl) {
#pragma unroll
        for (int d = 0; d < 8; ++d) {
            const int row = mt * 16 + d + 8 * h;
            const int col = c0 + tl * 16 + cl;
            out[(long)row * NBATCH + col] = acc[tl][d];
        }
    }
}

// ---------------------------------------------------------------------------
extern "C" void kernel_launch(void* const* d_in, const int* in_sizes, int n_in,
                              void* d_out, int out_size, void* d_ws, size_t ws_size,
                              hipStream_t stream) {
    const float* x = (const float*)d_in[0];   // [128, 16384]
    const float* A = (const float*)d_in[1];   // [128, 128]
    const float* B = (const float*)d_in[2];   // [128, 8256]
    float* out = (float*)d_out;               // [128, 16384]

    unsigned int*   rc = (unsigned int*)((char*)d_ws + WS_RC_OFF);
    unsigned short* wh = (unsigned short*)((char*)d_ws + WS_WH_OFF);
    unsigned short* wl = (unsigned short*)((char*)d_ws + WS_WL_OFF);

    build_rc_kernel<<<dim3(NROWS), dim3(128), 0, stream>>>(rc);
    pack_w_kernel<<<dim3(WPK_ELEMS / 256), dim3(256), 0, stream>>>(A, B, wh, wl);
    infs_main_kernel<<<dim3(NBATCH / 64), dim3(256), 0, stream>>>(
        x, rc, (const uint4*)wh, (const uint4*)wl, out);
}
